// TransformerLayer_72859825210064
// MI455X (gfx1250) — matrix operand order repaired and checked by reading, not compiled, not run on hardware
//
#include <hip/hip_runtime.h>
#include <math.h>

// ---------------------------------------------------------------------------
// Transformer layer for MI455X (gfx1250, wave32, WMMA).
// Compute path: bf16 WMMA (V_WMMA_F32_16X16X32_BF16) with fp32 accumulation,
// fp32 residuals / layernorm / softmax statistics.
// GEMM: each wave owns a 32x64 C tile (2x4 subtiles of 16x16), block = 8 waves
// arranged 4(M) x 2(N) covering a 128x128 C tile.
// ---------------------------------------------------------------------------

#define T_DIM   4096
#define DM      1024
#define DI      1024
#define DFF     4096
#define LN_EPS  1e-5f

typedef __bf16 bf16;
typedef __attribute__((ext_vector_type(16))) __bf16 v16bf;
typedef __attribute__((ext_vector_type(8)))  __bf16 v8bf;
typedef __attribute__((ext_vector_type(8)))  float  v8f;

// ---------------------------------------------------------------------------
// Pack a [Kd x Nd] matrix into WMMA B-fragment layout (bf16).
// Fragment f = n_tile * (Kd/32) + k_tile. Per lane (0..31):
//   n     = n_tile*16 + (lane & 15)
//   kbase = k_tile*32 + 8*(lane>>4)
//   16 bf16 elements e=0..15 hold B[kbase+e+(e>=8?8:0)][n], stored contiguously (32B).
// TRANS: B[k][n] = src[n*ld + k]   (used for K^T; reads are contiguous)
// else : B[k][n] = src[k*ld + n]
// ---------------------------------------------------------------------------
template <typename SRCT, bool TRANS>
__global__ __launch_bounds__(256) void pack_b_kernel(
    const SRCT* __restrict__ src, bf16* __restrict__ dst,
    int Kd, int Nd, int ld)
{
    int gid  = blockIdx.x * 256 + threadIdx.x;
    int lane = gid & 31;
    int f    = gid >> 5;
    int ktiles = Kd >> 5;
    int total  = ktiles * (Nd >> 4);
    if (f >= total) return;
    int nt = f / ktiles;
    int kt = f - nt * ktiles;
    int n  = (nt << 4) + (lane & 15);
    int kb = (kt << 5) + ((lane >> 4) << 3);
    bf16* d = dst + (((size_t)f << 5) + (size_t)lane) * 16;
#pragma unroll
    for (int e = 0; e < 16; ++e) {
        float v;
        if (TRANS) v = (float)src[(size_t)n * ld + (kb + e + ((e >> 3) << 3))];
        else       v = (float)src[(size_t)(kb + e + ((e >> 3) << 3)) * ld + n];
        d[e] = (bf16)v;
    }
}

// ---------------------------------------------------------------------------
// Row layernorm: one block per row, fp32 stats, bf16 output.
// ---------------------------------------------------------------------------
__global__ __launch_bounds__(256) void layernorm_bf16_kernel(
    const float* __restrict__ x, const float* __restrict__ g,
    const float* __restrict__ b, bf16* __restrict__ out, int D)
{
    __shared__ float s1[256];
    __shared__ float s2[256];
    int row = blockIdx.x;
    int tid = threadIdx.x;
    const float* p = x + (size_t)row * D;

    float a = 0.f, c = 0.f;
    for (int i = tid; i < D; i += 256) { float v = p[i]; a += v; c += v * v; }
    s1[tid] = a; s2[tid] = c;
    __syncthreads();
    for (int s = 128; s > 0; s >>= 1) {
        if (tid < s) { s1[tid] += s1[tid + s]; s2[tid] += s2[tid + s]; }
        __syncthreads();
    }
    float mean = s1[0] / (float)D;
    float var  = s2[0] / (float)D - mean * mean;
    float inv  = rsqrtf(var + LN_EPS);
    bf16* o = out + (size_t)row * D;
    for (int i = tid; i < D; i += 256)
        o[i] = (bf16)((p[i] - mean) * inv * g[i] + b[i]);
}

// ---------------------------------------------------------------------------
// Row softmax over T columns, in place (fp32) + bf16 copy for the next GEMM.
// ---------------------------------------------------------------------------
__global__ __launch_bounds__(256) void softmax_row_kernel(
    float* __restrict__ attn, bf16* __restrict__ attnB, int T)
{
    __shared__ float red[256];
    int row = blockIdx.x;
    int tid = threadIdx.x;
    float* p = attn + (size_t)row * T;

    float mx = -3.0e38f;
    for (int i = tid; i < T; i += 256) mx = fmaxf(mx, p[i]);
    red[tid] = mx;
    __syncthreads();
    for (int s = 128; s > 0; s >>= 1) {
        if (tid < s) red[tid] = fmaxf(red[tid], red[tid + s]);
        __syncthreads();
    }
    mx = red[0];
    __syncthreads();

    float sum = 0.f;
    for (int i = tid; i < T; i += 256) {
        float e = expf(p[i] - mx);
        p[i] = e;
        sum += e;
    }
    red[tid] = sum;
    __syncthreads();
    for (int s = 128; s > 0; s >>= 1) {
        if (tid < s) red[tid] += red[tid + s];
        __syncthreads();
    }
    float inv = 1.0f / red[0];
    bf16* ob = attnB + (size_t)row * T;
    for (int i = tid; i < T; i += 256) {
        float v = p[i] * inv;
        p[i]  = v;
        ob[i] = (bf16)v;
    }
}

// ---------------------------------------------------------------------------
// Register-blocked WMMA GEMM:
//   C[M,N] = act(scale * (A @ B) + bias) + residual
//   A : bf16 row-major [M,K];  Bp : bf16 pre-packed B-fragments.
// Block = 256 threads = 8 waves arranged 4(M) x 2(N); block tile 128x128.
// Wave tile = 32x64: 2 M-subtiles x 4 N-subtiles, 8 v8f accumulators.
// Per K-step (32): 2 A fragments + 4 B fragments feed 8 WMMAs
//   -> A reused 4x, B reused 2x, 0.75 fragment loads per WMMA.
// A fragment per lane: row m; two contiguous 16B runs at
//   K = kt*32 + 8*(lane>>4) + {0..7} and +16..+23  (ISA 16-bit A layout).
// C layout: acc[r] -> (m = msub + 8*(lane>>4) + r, n = nsub + (lane&15)).
// ---------------------------------------------------------------------------
__global__ __launch_bounds__(256) void gemm_wmma_bf16_kernel(
    const bf16* __restrict__ A, const bf16* __restrict__ Bp,
    const float* __restrict__ bias, const float* __restrict__ residual,
    float* __restrict__ outF, bf16* __restrict__ outB,
    int M, int N, int K, float scale, int act)
{
    const int lane = threadIdx.x & 31;
    const int wave = threadIdx.x >> 5;
    const int wm   = wave >> 1;              // 0..3 : M group within block
    const int wn   = wave & 1;               // 0..1 : N group within block
    const int m0   = blockIdx.y * 128 + wm * 32;
    const int n0   = blockIdx.x * 128 + wn * 64;
    if (m0 >= M || n0 >= N) return;

    const int khalf  = lane >> 4;
    const int ktiles = K >> 5;

    // A rows for the two M-subtiles
    const bf16* arow0 = A + (size_t)(m0 + (lane & 15)) * K;
    const bf16* arow1 = arow0 + (size_t)16 * K;

    // Packed-B pointers for the four N-subtiles (advance 32 v16bf per K-step)
    const int nt_base = (n0 >> 4);
    const v16bf* bptr0 = (const v16bf*)Bp + (size_t)(nt_base + 0) * ktiles * 32 + lane;
    const v16bf* bptr1 = (const v16bf*)Bp + (size_t)(nt_base + 1) * ktiles * 32 + lane;
    const v16bf* bptr2 = (const v16bf*)Bp + (size_t)(nt_base + 2) * ktiles * 32 + lane;
    const v16bf* bptr3 = (const v16bf*)Bp + (size_t)(nt_base + 3) * ktiles * 32 + lane;

    v8f acc[2][4];
#pragma unroll
    for (int i = 0; i < 2; ++i)
#pragma unroll
        for (int j = 0; j < 4; ++j) acc[i][j] = (v8f){};

    for (int kt = 0; kt < ktiles; ++kt) {
        const int kb0 = (kt << 5) + (khalf << 3);

        v16bf a0, a1;
        {
            v8bf lo = *(const v8bf*)(arow0 + kb0);
            v8bf hi = *(const v8bf*)(arow0 + kb0 + 16);
#pragma unroll
            for (int e = 0; e < 8; ++e) { a0[e] = lo[e]; a0[8 + e] = hi[e]; }
        }
        {
            v8bf lo = *(const v8bf*)(arow1 + kb0);
            v8bf hi = *(const v8bf*)(arow1 + kb0 + 16);
#pragma unroll
            for (int e = 0; e < 8; ++e) { a1[e] = lo[e]; a1[8 + e] = hi[e]; }
        }

        v16bf b0 = bptr0[(size_t)kt * 32];
        v16bf b1 = bptr1[(size_t)kt * 32];
        v16bf b2 = bptr2[(size_t)kt * 32];
        v16bf b3 = bptr3[(size_t)kt * 32];

#if defined(__gfx1250__)
        acc[0][0] = __builtin_amdgcn_wmma_f32_16x16x32_bf16(false, a0, false, b0, (short)0, acc[0][0], false, false);
        acc[0][1] = __builtin_amdgcn_wmma_f32_16x16x32_bf16(false, a0, false, b1, (short)0, acc[0][1], false, false);
        acc[0][2] = __builtin_amdgcn_wmma_f32_16x16x32_bf16(false, a0, false, b2, (short)0, acc[0][2], false, false);
        acc[0][3] = __builtin_amdgcn_wmma_f32_16x16x32_bf16(false, a0, false, b3, (short)0, acc[0][3], false, false);
        acc[1][0] = __builtin_amdgcn_wmma_f32_16x16x32_bf16(false, a1, false, b0, (short)0, acc[1][0], false, false);
        acc[1][1] = __builtin_amdgcn_wmma_f32_16x16x32_bf16(false, a1, false, b1, (short)0, acc[1][1], false, false);
        acc[1][2] = __builtin_amdgcn_wmma_f32_16x16x32_bf16(false, a1, false, b2, (short)0, acc[1][2], false, false);
        acc[1][3] = __builtin_amdgcn_wmma_f32_16x16x32_bf16(false, a1, false, b3, (short)0, acc[1][3], false, false);
#else
        acc[0][0][0] += (float)a0[0] * (float)b0[0];  // host / non-gfx1250 parse path
        acc[1][3][0] += (float)a1[0] * (float)b3[0];
        (void)b1; (void)b2;
#endif
    }

    // Epilogue: bias / scale / gelu / residual, fp32 and/or bf16 stores.
#pragma unroll
    for (int j = 0; j < 4; ++j) {
        const int n  = n0 + j * 16 + (lane & 15);
        const float bn = bias ? bias[n] : 0.0f;
#pragma unroll
        for (int i = 0; i < 2; ++i) {
#pragma unroll
            for (int r = 0; r < 8; ++r) {
                const int m = m0 + i * 16 + (khalf << 3) + r;
                float v = acc[i][j][r] * scale + bn;
                if (act == 1)                                    // exact GELU
                    v = 0.5f * v * (1.0f + erff(v * 0.70710678118654752f));
                if (residual) v += residual[(size_t)m * N + n];
                const size_t idx = (size_t)m * N + n;
                if (outF) outF[idx] = v;
                if (outB) outB[idx] = (bf16)v;
            }
        }
    }
}

// ---------------------------------------------------------------------------
// Host-side launcher
// ---------------------------------------------------------------------------
static inline int cdiv(int a, int b) { return (a + b - 1) / b; }

extern "C" void kernel_launch(void* const* d_in, const int* in_sizes, int n_in,
                              void* d_out, int out_size, void* d_ws, size_t ws_size,
                              hipStream_t stream)
{
    (void)in_sizes; (void)n_in; (void)out_size; (void)ws_size;

    const float* input = (const float*)d_in[0];
    const float* Wq = (const float*)d_in[1];  const float* bq = (const float*)d_in[2];
    const float* Wk = (const float*)d_in[3];  const float* bk = (const float*)d_in[4];
    const float* Wv = (const float*)d_in[5];  const float* bv = (const float*)d_in[6];
    const float* Wo = (const float*)d_in[7];  const float* bo = (const float*)d_in[8];
    const float* W1 = (const float*)d_in[9];  const float* b1 = (const float*)d_in[10];
    const float* W2 = (const float*)d_in[11]; const float* b2 = (const float*)d_in[12];
    const float* g1 = (const float*)d_in[13]; const float* be1 = (const float*)d_in[14];
    const float* g2 = (const float*)d_in[15]; const float* be2 = (const float*)d_in[16];

    float* outX  = (float*)d_out;                       // [T, DM]
    float* attnF = (float*)d_out + (size_t)T_DIM * DM;  // [T, T]

    // Workspace carve-up (bytes)
    char* ws = (char*)d_ws;
    size_t o = 0;
    auto carve = [&](size_t bytes) { char* p = ws + o; o += bytes; return p; };
    bf16* WqP  = (bf16*)carve((size_t)DM  * DI  * 2);
    bf16* WkP  = (bf16*)carve((size_t)DM  * DI  * 2);
    bf16* WvP  = (bf16*)carve((size_t)DM  * DI  * 2);
    bf16* WoP  = (bf16*)carve((size_t)DI  * DM  * 2);
    bf16* W1P  = (bf16*)carve((size_t)DM  * DFF * 2);
    bf16* W2P  = (bf16*)carve((size_t)DFF * DM  * 2);
    bf16* xln  = (bf16*)carve((size_t)T_DIM * DM * 2);
    bf16* Qb   = (bf16*)carve((size_t)T_DIM * DI * 2);
    bf16* Kb   = (bf16*)carve((size_t)T_DIM * DI * 2);
    bf16* Vb   = (bf16*)carve((size_t)T_DIM * DI * 2);
    bf16* KtP  = (bf16*)carve((size_t)DI * T_DIM * 2);   // packed K^T  [DI x T]
    bf16* VP   = (bf16*)carve((size_t)T_DIM * DI * 2);   // packed V    [T x DI]
    bf16* attnB= (bf16*)carve((size_t)T_DIM * T_DIM * 2);
    bf16* ctxB = (bf16*)carve((size_t)T_DIM * DI * 2);
    float* xres= (float*)carve((size_t)T_DIM * DM * 4);
    bf16* ln2b = (bf16*)carve((size_t)T_DIM * DM * 2);
    bf16* hB   = (bf16*)carve((size_t)T_DIM * DFF * 2);

    auto pack_threads = [](int Kd, int Nd) { return (Kd >> 5) * (Nd >> 4) * 32; };

    // --- pack all weights to WMMA B-fragment layout -----------------------
    pack_b_kernel<float, false><<<cdiv(pack_threads(DM, DI), 256), 256, 0, stream>>>(Wq, WqP, DM, DI, DI);
    pack_b_kernel<float, false><<<cdiv(pack_threads(DM, DI), 256), 256, 0, stream>>>(Wk, WkP, DM, DI, DI);
    pack_b_kernel<float, false><<<cdiv(pack_threads(DM, DI), 256), 256, 0, stream>>>(Wv, WvP, DM, DI, DI);
    pack_b_kernel<float, false><<<cdiv(pack_threads(DI, DM), 256), 256, 0, stream>>>(Wo, WoP, DI, DM, DM);
    pack_b_kernel<float, false><<<cdiv(pack_threads(DM, DFF), 256), 256, 0, stream>>>(W1, W1P, DM, DFF, DFF);
    pack_b_kernel<float, false><<<cdiv(pack_threads(DFF, DM), 256), 256, 0, stream>>>(W2, W2P, DFF, DM, DM);

    // --- LN1 ---------------------------------------------------------------
    layernorm_bf16_kernel<<<T_DIM, 256, 0, stream>>>(input, g1, be1, xln, DM);

    // --- Q, K, V -----------------------------------------------------------
    dim3 gQKV(DI / 128, T_DIM / 128);
    gemm_wmma_bf16_kernel<<<gQKV, 256, 0, stream>>>(xln, WqP, bq, nullptr, nullptr, Qb, T_DIM, DI, DM, 1.0f, 0);
    gemm_wmma_bf16_kernel<<<gQKV, 256, 0, stream>>>(xln, WkP, bk, nullptr, nullptr, Kb, T_DIM, DI, DM, 1.0f, 0);
    gemm_wmma_bf16_kernel<<<gQKV, 256, 0, stream>>>(xln, WvP, bv, nullptr, nullptr, Vb, T_DIM, DI, DM, 1.0f, 0);

    // --- repack K^T and V for the attention GEMMs --------------------------
    pack_b_kernel<bf16, true ><<<cdiv(pack_threads(DI, T_DIM), 256), 256, 0, stream>>>(Kb, KtP, DI, T_DIM, DI);
    pack_b_kernel<bf16, false><<<cdiv(pack_threads(T_DIM, DI), 256), 256, 0, stream>>>(Vb, VP, T_DIM, DI, DI);

    // --- scores = Q @ K^T / sqrt(DI) --------------------------------------
    dim3 gScore(T_DIM / 128, T_DIM / 128);
    gemm_wmma_bf16_kernel<<<gScore, 256, 0, stream>>>(Qb, KtP, nullptr, nullptr, attnF, nullptr,
                                                      T_DIM, T_DIM, DI, 0.03125f /*1/sqrt(1024)*/, 0);

    // --- softmax (fp32 in place -> d_out attn region; bf16 copy) ----------
    softmax_row_kernel<<<T_DIM, 256, 0, stream>>>(attnF, attnB, T_DIM);

    // --- context = attn @ V ------------------------------------------------
    dim3 gCtx(DI / 128, T_DIM / 128);
    gemm_wmma_bf16_kernel<<<gCtx, 256, 0, stream>>>(attnB, VP, nullptr, nullptr, nullptr, ctxB,
                                                    T_DIM, DI, T_DIM, 1.0f, 0);

    // --- x = input + context @ Wo + bo  (fp32 residual) --------------------
    gemm_wmma_bf16_kernel<<<gCtx, 256, 0, stream>>>(ctxB, WoP, bo, input, xres, nullptr,
                                                    T_DIM, DM, DI, 1.0f, 0);

    // --- LN2 ---------------------------------------------------------------
    layernorm_bf16_kernel<<<T_DIM, 256, 0, stream>>>(xres, g2, be2, ln2b, DM);

    // --- h = gelu(ln2 @ W1 + b1) -------------------------------------------
    dim3 gFF1(DFF / 128, T_DIM / 128);
    gemm_wmma_bf16_kernel<<<gFF1, 256, 0, stream>>>(ln2b, W1P, b1, nullptr, nullptr, hB,
                                                    T_DIM, DFF, DM, 1.0f, 1);

    // --- out = x + h @ W2 + b2  --------------------------------------------
    dim3 gFF2(DM / 128, T_DIM / 128);
    gemm_wmma_bf16_kernel<<<gFF2, 256, 0, stream>>>(hB, W2P, b2, xres, outX, nullptr,
                                                    T_DIM, DM, DFF, 1.0f, 0);
}
